// Attention_Layer_30039001268548
// MI455X (gfx1250) — compile-verified, matrix operand order checked by reading, not run on hardware
//
#include <hip/hip_runtime.h>
#include <hip/hip_bf16.h>

// ---------------------------------------------------------------------------
// Attention layer: B=8, S=4096, D=128, fp32 in/out.
//   Q=xWq^T, K=xWk^T, V=xWv^T ; out = softmax(QK^T) V   (unscaled scores)
//
// MI455X strategy:
//  - QKV projections: memory-bound -> one pass, f16 WMMA, outputs written
//    directly in WMMA per-lane fragment layouts (A-frags for Q, B-frags for
//    K and V) so the attention hot loop is pure fragment traffic.
//  - Attention: ~69 GFLOP compute-bound -> flash attention.  8 waves per
//    workgroup share each 32-key K/V fragment chunk staged into LDS by the
//    Tensor Data Mover (double-buffered, TENSORcnt-tracked), giving 8x less
//    L2 traffic than per-wave streaming.  Softmax stats in f32 with wave32
//    shfl_xor butterflies; P goes C-layout -> A-layout through per-wave LDS
//    (ordered by s_wait_dscnt).
// ---------------------------------------------------------------------------

typedef __attribute__((ext_vector_type(16))) _Float16 v16h;
typedef __attribute__((ext_vector_type(8)))  _Float16 v8h;
typedef __attribute__((ext_vector_type(8)))  float    v8f;
typedef __attribute__((ext_vector_type(4)))  unsigned v4u;
typedef __attribute__((ext_vector_type(8)))  int      v8i;
typedef __attribute__((ext_vector_type(4)))  int      v4i;

#define DIM   128
#define SEQ   4096
#define BATCH 8
#define NKC   (SEQ / 32)        // 128 key chunks of 32 keys

#if defined(__gfx1250__) && __has_builtin(__builtin_amdgcn_tensor_load_to_lds) && \
    __has_builtin(__builtin_amdgcn_s_wait_tensorcnt)
#define USE_TDM 1
#else
#define USE_TDM 0
#endif

static __device__ __forceinline__ v8f wmma_f16(v16h a, v16h b, v8f c) {
  // D = A(16x32 f16) * B(32x16 f16) + C(16x16 f32)
  return __builtin_amdgcn_wmma_f32_16x16x32_f16(
      false, a, false, b, (short)0, c, false, false);
}

#if USE_TDM
// TDM: copy one contiguous 8 KB chunk (1024 x 8B elements, 1-D tile) from
// global memory into LDS.  D# fields per CDNA5 ISA sec.8.
static __device__ __forceinline__ void tdm_copy_8k(const void* gsrc, void* ldst) {
  unsigned lds_addr = (unsigned)(unsigned long long)ldst;     // addr[31:0] = LDS offset
  unsigned long long ga = (unsigned long long)gsrc;
  // group0: count=1 | lds_addr | global_addr[56:0] | type=2
  v4u g0 = { 1u,
             lds_addr,
             (unsigned)ga,
             (unsigned)((ga >> 32) & 0x01FFFFFFu) | (2u << 30) };
  const unsigned elems = 1024;                                 // 8KB / 8B
  // group1: data_size=3 (8B); tensor_dim0=1024; tensor_dim1=1; tile_dim0=1024
  v8i g1 = { (int)(3u << 16),
             (int)(elems << 16),            // tensor_dim0 -> bits[79:48]
             (int)(1u << 16),               // tensor_dim1 -> bits[111:80]
             (int)(elems << 16),            // tile_dim0   -> bits[127:112]
             0,
             (int)elems,                    // tensor_dim0_stride -> bits[207:160]
             0, 0 };
  v4i z4 = {0, 0, 0, 0};
#if __clang_major__ >= 23
  v8i z8 = {0, 0, 0, 0, 0, 0, 0, 0};
  __builtin_amdgcn_tensor_load_to_lds(g0, g1, z4, z4, z8, 0);
#else
  __builtin_amdgcn_tensor_load_to_lds(g0, g1, z4, z4, 0);
#endif
  asm volatile("" ::: "memory");
}
#endif

// Fragment layout conventions (wave32, 16x16x32 f16 WMMA):
//  A (16x32, MxK):  lane = m + 16*hi,  hi = (k%32)/8 & 1, idx = (k%8)+8*((k%32)/16)
//  B (32x16, KxN):  lane = n + 16*hi,  same k packing
//  C (16x16 f32):   lane = n + 16*(m/8), vgpr r = m%8
// Fragments live in workspace as [tile][lane][16 halves] -> 32B/lane loads.

// ---------------------------------------------------------------------------
// Kernel 0: pack Wq/Wk/Wv (fp32 [e][d]) into f16 B-fragments.
// ---------------------------------------------------------------------------
__global__ void pack_weights_kernel(const float* __restrict__ Wq,
                                    const float* __restrict__ Wk,
                                    const float* __restrict__ Wv,
                                    _Float16* __restrict__ wf) {
  int idx = blockIdx.x * blockDim.x + threadIdx.x;   // 0 .. 3*128*128-1
  int which = idx >> 14;
  int rem   = idx & 16383;
  int e = rem >> 7;
  int d = rem & 127;
  const float* W = (which == 0) ? Wq : (which == 1) ? Wk : Wv;
  float val = W[rem];

  int ec  = d >> 5;
  int d32 = d & 31;
  int et  = e >> 4;
  int n   = e & 15;
  int lane = n + 16 * ((d32 >> 3) & 1);
  int h    = (d32 & 7) + 8 * (d32 >> 4);
  wf[(((which * 4 + ec) * 8 + et) * 32 + lane) * 16 + h] = (_Float16)val;
}

// ---------------------------------------------------------------------------
// Kernel 1: QKV projection.  One wave per 16-row tile (grid = B*S/16).
//   Q as A-fragments [tile][ec(4)][lane][16]
//   K as B-fragments [tile][ec(4)][lane][16]   (same bytes as A of K)
//   V as B-fragments [b][kc(128)][et(8)][lane][16]
// ---------------------------------------------------------------------------
__global__ void qkv_proj_kernel(const float* __restrict__ x,
                                const _Float16* __restrict__ wf,
                                _Float16* __restrict__ qf,
                                _Float16* __restrict__ kf,
                                _Float16* __restrict__ vf) {
  const int lane = threadIdx.x & 31;
  const int tile = blockIdx.x;              // b*256 + st
  const int n    = lane & 15;
  const int half = lane >> 4;
  const long row = (long)tile * 16 + n;     // A-layout row m = lane%16

  const v16h* wfv = (const v16h*)wf;

  v16h a[4];
  #pragma unroll
  for (int ec = 0; ec < 4; ++ec) {
    const float* p = x + row * DIM + ec * 32 + half * 8;
    float4 f0 = *(const float4*)(p);
    float4 f1 = *(const float4*)(p + 4);
    float4 f2 = *(const float4*)(p + 16);
    float4 f3 = *(const float4*)(p + 20);
    v16h t;
    t[0]=(_Float16)f0.x;  t[1]=(_Float16)f0.y;  t[2]=(_Float16)f0.z;  t[3]=(_Float16)f0.w;
    t[4]=(_Float16)f1.x;  t[5]=(_Float16)f1.y;  t[6]=(_Float16)f1.z;  t[7]=(_Float16)f1.w;
    t[8]=(_Float16)f2.x;  t[9]=(_Float16)f2.y;  t[10]=(_Float16)f2.z; t[11]=(_Float16)f2.w;
    t[12]=(_Float16)f3.x; t[13]=(_Float16)f3.y; t[14]=(_Float16)f3.z; t[15]=(_Float16)f3.w;
    a[ec] = t;
  }

  #pragma unroll
  for (int which = 0; which < 3; ++which) {
    _Float16* dst = (which == 0) ? qf : (which == 1) ? kf : vf;
    #pragma unroll
    for (int et = 0; et < 8; ++et) {
      v8f c = {0.f,0.f,0.f,0.f,0.f,0.f,0.f,0.f};
      #pragma unroll
      for (int ec = 0; ec < 4; ++ec)
        c = wmma_f16(a[ec], wfv[((which * 4 + ec) * 8 + et) * 32 + lane], c);

      // Scatter C-layout (row m=r+8*half, col n) into fragment buffers.
      #pragma unroll
      for (int r = 0; r < 8; ++r) {
        int m_row = r + 8 * half;
        long flat;
        if (which < 2) {
          flat = (((long)tile * 4 + (et >> 1)) * 32 + m_row + 16 * (n >> 3)) * 16
               + (n & 7) + 8 * (et & 1);
        } else {
          int s_in_b = ((tile & 255) << 4) + m_row;
          int bb  = tile >> 8;
          int kc  = s_in_b >> 5;
          int s32 = s_in_b & 31;
          flat = ((((long)bb * 128 + kc) * 8 + et) * 32
                  + (n + 16 * ((s32 >> 3) & 1))) * 16
               + (s32 & 7) + 8 * (s32 >> 4);
        }
        dst[flat] = (_Float16)c[r];
      }
    }
  }
}

// ---------------------------------------------------------------------------
// Kernel 2: flash attention.  256 threads = 8 waves; each wave owns one
// 16-query tile; the workgroup shares TDM-staged K/V chunks in LDS.
// Grid = B*S/128 = 256 blocks.
// ---------------------------------------------------------------------------
__global__ void __launch_bounds__(256)
flash_attn_kernel(const _Float16* __restrict__ qf,
                  const _Float16* __restrict__ kf,
                  const _Float16* __restrict__ vf,
                  float* __restrict__ out) {
  __shared__ _Float16 lds_k[2][2 * 4 * 32 * 16];   // double-buffered K chunk (2x 8KB)
  __shared__ _Float16 lds_v[2][8 * 32 * 16];       // double-buffered V chunk (2x 8KB)
  __shared__ _Float16 lds_p[8][16 * 32];           // per-wave P tile (8x 1KB)

  const int tid  = threadIdx.x;
  const int lane = tid & 31;
  const int wave = tid >> 5;
  const int blk  = blockIdx.x;
  const int b    = blk >> 5;                       // 32 blocks per batch
  const int tile = b * 256 + (blk & 31) * 8 + wave; // global 16-query tile
  const int n16  = lane & 15;
  const int half = lane >> 4;

  const v16h* qv = (const v16h*)qf;

  // Stage one 32-key chunk (K: 2 tiles x 4 ec; V: 8 et) into LDS buffer.
  auto stage = [&](int kc, int buf) {
    const _Float16* ksrc = kf + ((long)(b * 256 + kc * 2) * 4 * 32) * 16;
    const _Float16* vsrc = vf + (((long)b * 128 + kc) * 8 * 32) * 16;
#if USE_TDM
    if (wave == 0) {
      tdm_copy_8k(ksrc, &lds_k[buf][0]);
      tdm_copy_8k(vsrc, &lds_v[buf][0]);
    }
#else
    ((v16h*)&lds_k[buf][0])[tid] = ((const v16h*)ksrc)[tid];
    ((v16h*)&lds_v[buf][0])[tid] = ((const v16h*)vsrc)[tid];
#endif
  };

  // Q tile: 4 resident A-fragments.
  v16h aq[4];
  #pragma unroll
  for (int ec = 0; ec < 4; ++ec)
    aq[ec] = qv[((long)tile * 4 + ec) * 32 + lane];

  const v8f zero8 = {0.f,0.f,0.f,0.f,0.f,0.f,0.f,0.f};
  v8f acc[8];
  #pragma unroll
  for (int et = 0; et < 8; ++et) acc[et] = zero8;
  float mrow[8], lrow[8];
  #pragma unroll
  for (int r = 0; r < 8; ++r) { mrow[r] = -1e30f; lrow[r] = 0.f; }

  stage(0, 0);

  for (int kc = 0; kc < NKC; ++kc) {
    const int buf = kc & 1;
    if (kc + 1 < NKC) stage(kc + 1, buf ^ 1);
#if USE_TDM
    if (wave == 0) {
      if (kc + 1 < NKC) __builtin_amdgcn_s_wait_tensorcnt(1);  // buf done, next in flight
      else              __builtin_amdgcn_s_wait_tensorcnt(0);
    }
#endif
    __syncthreads();                                 // buf visible to all waves

    const v16h* lk = (const v16h*)&lds_k[buf][0];
    const v16h* lv = (const v16h*)&lds_v[buf][0];

    // ---- S = Q K^T : two 16x16 score tiles over the 32 keys ----
    v8f s0 = zero8, s1 = zero8;
    #pragma unroll
    for (int ec = 0; ec < 4; ++ec)
      s0 = wmma_f16(aq[ec], lk[(0 * 4 + ec) * 32 + lane], s0);
    #pragma unroll
    for (int ec = 0; ec < 4; ++ec)
      s1 = wmma_f16(aq[ec], lk[(1 * 4 + ec) * 32 + lane], s1);

    // ---- online softmax: rows in vgpr r, cols in lanes%16 ----
    float sc[8];
    #pragma unroll
    for (int r = 0; r < 8; ++r) {
      float rmax = fmaxf(s0[r], s1[r]);
      #pragma unroll
      for (int mask = 1; mask <= 8; mask <<= 1)
        rmax = fmaxf(rmax, __shfl_xor(rmax, mask, 32));  // stays in 16-lane half
      float mn = fmaxf(mrow[r], rmax);
      sc[r]    = __expf(mrow[r] - mn);
      float p0 = __expf(s0[r] - mn);
      float p1 = __expf(s1[r] - mn);
      float rs = p0 + p1;
      #pragma unroll
      for (int mask = 1; mask <= 8; mask <<= 1)
        rs += __shfl_xor(rs, mask, 32);
      lrow[r] = lrow[r] * sc[r] + rs;
      mrow[r] = mn;
      lds_p[wave][(r + 8 * half) * 32 + n16]      = (_Float16)p0;
      lds_p[wave][(r + 8 * half) * 32 + 16 + n16] = (_Float16)p1;
    }
    #pragma unroll
    for (int et = 0; et < 8; ++et) {
      #pragma unroll
      for (int r = 0; r < 8; ++r) acc[et][r] *= sc[r];
    }

    // in-wave LDS RAW fence (CDNA5 split counters)
    asm volatile("s_wait_dscnt 0x0" ::: "memory");

    // ---- P reloaded as A-fragment: lane m=n16, k set {8h..8h+7, +16} ----
    v8h plo = *(const v8h*)&lds_p[wave][n16 * 32 + 8 * half];
    v8h phi = *(const v8h*)&lds_p[wave][n16 * 32 + 8 * half + 16];
    v16h pf;
    #pragma unroll
    for (int i = 0; i < 8; ++i) { pf[i] = plo[i]; pf[i + 8] = phi[i]; }

    // ---- O += P V ----
    #pragma unroll
    for (int et = 0; et < 8; ++et)
      acc[et] = wmma_f16(pf, lv[et * 32 + lane], acc[et]);

    __syncthreads();                                 // all waves done with buf
  }

  // ---- epilogue: out[b][s][e] = acc / l ----
  const long ob = (long)tile * 16 * DIM;
  #pragma unroll
  for (int et = 0; et < 8; ++et) {
    #pragma unroll
    for (int r = 0; r < 8; ++r) {
      out[ob + (long)(r + 8 * half) * DIM + et * 16 + n16] = acc[et][r] / lrow[r];
    }
  }
}

// ---------------------------------------------------------------------------
extern "C" void kernel_launch(void* const* d_in, const int* in_sizes, int n_in,
                              void* d_out, int out_size, void* d_ws, size_t ws_size,
                              hipStream_t stream) {
  (void)in_sizes; (void)n_in; (void)out_size; (void)ws_size;

  const float* x  = (const float*)d_in[0];   // (8, 4096, 128)
  const float* Wq = (const float*)d_in[1];   // (128, 128)
  const float* Wk = (const float*)d_in[2];
  const float* Wv = (const float*)d_in[3];
  float* out = (float*)d_out;                // (8, 4096, 128)

  // Workspace: wf 96KiB (padded to 128KiB), qf/kf/vf 8MiB each (~24.2MiB).
  char* ws = (char*)d_ws;
  _Float16* wf = (_Float16*)(ws);
  _Float16* qf = (_Float16*)(ws + (128 << 10));
  _Float16* kf = (_Float16*)(ws + (128 << 10) + ((size_t)8 << 20));
  _Float16* vf = (_Float16*)(ws + (128 << 10) + ((size_t)16 << 20));

  pack_weights_kernel<<<192, 256, 0, stream>>>(Wq, Wk, Wv, wf);
  qkv_proj_kernel<<<(BATCH * SEQ) / 16, 32, 0, stream>>>(x, wf, qf, kf, vf);
  flash_attn_kernel<<<(BATCH * SEQ) / 128, 256, 0, stream>>>(qf, kf, vf, out);
}
